// ParaNet_Point_78323023610164
// MI455X (gfx1250) — matrix operand order repaired and checked by reading, not compiled, not run on hardware
//
#include <hip/hip_runtime.h>
#include <math.h>

typedef __attribute__((ext_vector_type(16))) __bf16  v16bf;
typedef __attribute__((ext_vector_type(8)))  __bf16  v8bf;
typedef __attribute__((ext_vector_type(8)))  float   v8f;

#define PRADIUS 0.0125f   // 5 * 0.0025
#define EPSF    1e-12f

__device__ __forceinline__ float bsign(float x) {
    return (x > 0.f) ? 1.f : ((x < 0.f) ? -1.f : 0.f);
}

// ---------------- BatchNorm statistics ----------------
__global__ void stats_zero(float* s) { if (threadIdx.x < 4) s[threadIdx.x] = 0.f; }

__global__ void stats_acc(const float* __restrict__ vel, float* __restrict__ s, int N) {
    __shared__ float red[4][256];
    float sx = 0.f, sy = 0.f, sxx = 0.f, syy = 0.f;
    for (int n = blockIdx.x * blockDim.x + threadIdx.x; n < N; n += gridDim.x * blockDim.x) {
        float vx = vel[2*n], vy = vel[2*n+1];
        sx += vx; sy += vy; sxx += vx*vx; syy += vy*vy;
    }
    int t = threadIdx.x;
    red[0][t] = sx; red[1][t] = sy; red[2][t] = sxx; red[3][t] = syy;
    __syncthreads();
    for (int off = 128; off > 0; off >>= 1) {
        if (t < off) {
            #pragma unroll
            for (int j = 0; j < 4; ++j) red[j][t] += red[j][t + off];
        }
        __syncthreads();
    }
    if (t == 0) {
        atomicAdd(&s[0], red[0][0]); atomicAdd(&s[1], red[1][0]);
        atomicAdd(&s[2], red[2][0]); atomicAdd(&s[3], red[3][0]);
    }
}

__global__ void bn_feats(const float* __restrict__ vel, const float* __restrict__ s,
                         const float* __restrict__ gamma, const float* __restrict__ beta,
                         float* __restrict__ feats, int N) {
    int n = blockIdx.x * blockDim.x + threadIdx.x;
    if (n >= N) return;
    float inv = 1.f / (float)N;
    float mx = s[0]*inv, my = s[1]*inv;
    float vx = s[2]*inv - mx*mx, vy = s[3]*inv - my*my;
    float ix = rsqrtf(vx + 1e-5f), iy = rsqrtf(vy + 1e-5f);
    feats[3*n+0] = gamma[0]*(vel[2*n+0]-mx)*ix + beta[0];
    feats[3*n+1] = gamma[1]*(vel[2*n+1]-my)*iy + beta[1];
    feats[3*n+2] = beta[2];            // z column: mean=var=0 -> beta
}

// ---------------- geometry: window * bilinear taps, reused by all 5 layers ----------------
__global__ void geom_kernel(const float* __restrict__ pos, const int* __restrict__ nbr,
                            const unsigned char* __restrict__ mask,
                            float* __restrict__ wfull, int N) {
    int i = blockIdx.x * blockDim.x + threadIdx.x;
    if (i >= N * 9) return;
    int n = i / 9;
    int id = nbr[i];
    float x = (pos[2*id+0] - pos[2*n+0]) * (1.f / PRADIUS);
    float y = (pos[2*id+1] - pos[2*n+1]) * (1.f / PRADIUS);
    float r2 = x*x + y*y;
    float tw = 1.f - fminf(fmaxf(r2, 0.f), 1.f);
    float win = fminf(fmaxf(tw*tw*tw, 0.f), 1.f);
    win *= ((r2 <= 1.f) && (mask[i] != 0)) ? 1.f : 0.f;
    // ball->cube with z == 0: sphere->cylinder collapses to identity on (x,y)
    float cx = (r2 > EPSF) ? x : 0.f;
    float cy = (r2 > EPSF) ? y : 0.f;
    float c2 = cx*cx + cy*cy;
    float r  = sqrtf(fmaxf(c2, EPSF));
    const float fop = 1.27323954473516268f;  // 4/pi
    float dx = (fabsf(cx) > 1e-9f) ? cx : 1.f;
    float dy = (fabsf(cy) > 1e-9f) ? cy : 1.f;
    float u, v;
    if (fabsf(cx) >= fabsf(cy)) {
        u = bsign(cx) * r;
        v = bsign(cx) * r * fop * atanf(cy / dx);
    } else {
        u = bsign(cy) * r * fop * atanf(cx / dy);
        v = bsign(cy) * r;
    }
    if (c2 <= EPSF) { u = 0.f; v = 0.f; }
    float gx = fminf(fmaxf((u + 1.f) * 1.5f, 0.f), 3.f);
    float gy = fminf(fmaxf((v + 1.f) * 1.5f, 0.f), 3.f);
    float wx[4], wy[4];
    #pragma unroll
    for (int t = 0; t < 4; ++t) {
        wx[t] = fmaxf(1.f - fabsf(gx - (float)t), 0.f);
        wy[t] = fmaxf(1.f - fabsf(gy - (float)t), 0.f);
    }
    #pragma unroll
    for (int a = 0; a < 4; ++a)
        #pragma unroll
        for (int b = 0; b < 4; ++b)
            wfull[(size_t)i * 16 + a * 4 + b] = wx[a] * wy[b] * win;
}

// ---------------- weight convert: f32 (G,C,D) -> bf16 (G,D,C)  (B-fragment friendly) ----------------
__global__ void w_to_bf16T(const float* __restrict__ src, __bf16* __restrict__ dst,
                           int G, int C, int D) {
    int i = blockIdx.x * blockDim.x + threadIdx.x;
    int total = G * C * D;
    if (i >= total) return;
    int g = i / (C * D); int rem = i - g * C * D; int c = rem / D; int d = rem - c * D;
    dst[((size_t)g * D + d) * C + c] = (__bf16)src[i];
}

// ---------------- layer 0 (Cin=3): plain VALU, trivial FLOPs ----------------
__global__ void layer0_kernel(const float* __restrict__ feats, const float* __restrict__ wfull,
                              const int* __restrict__ nbr,
                              const float* __restrict__ cw, const float* __restrict__ cb,
                              const float* __restrict__ dw, const float* __restrict__ db,
                              float* __restrict__ f0, int N) {
    int n = blockIdx.x * blockDim.x + threadIdx.x;
    if (n >= N) return;
    float t[48];
    #pragma unroll
    for (int j = 0; j < 48; ++j) t[j] = 0.f;
    #pragma unroll
    for (int k = 0; k < 9; ++k) {
        int id = nbr[n*9 + k];
        float fx = feats[3*id], fy = feats[3*id+1], fz = feats[3*id+2];
        #pragma unroll
        for (int g = 0; g < 16; ++g) {
            float w = wfull[(size_t)n*144 + k*16 + g];
            t[g*3+0] += w * fx; t[g*3+1] += w * fy; t[g*3+2] += w * fz;
        }
    }
    float s0 = feats[3*n], s1 = feats[3*n+1], s2 = feats[3*n+2];
    for (int d = 0; d < 32; ++d) {
        float acc = cb[d];
        #pragma unroll
        for (int j = 0; j < 48; ++j) acc += t[j] * cw[j*32 + d];
        f0[(size_t)n*64 + d] = acc;
    }
    for (int d = 0; d < 32; ++d)
        f0[(size_t)n*64 + 32 + d] = db[d] + s0*dw[d] + s1*dw[32+d] + s2*dw[64+d];
}

// ---------------- fused gather + WMMA conv + dense (+residual), layers 1..3 ----------------
template<int CIN, int CO, bool RES>
__global__ __launch_bounds__(256)
void conv_layer_kernel(const float* __restrict__ fin,        // N x CIN, pre-activation
                       const __bf16* __restrict__ wB,        // bf16 (16,CO,CIN)
                       const float* __restrict__ cb,
                       const __bf16* __restrict__ wD,        // bf16 (CO,CIN)
                       const float* __restrict__ db,
                       const float* __restrict__ wfull,      // N x 144
                       const int* __restrict__ nbr,          // N x 9
                       float* __restrict__ fout)             // N x CO
{
    constexpr int MT  = 32;                    // particles per block (2 WMMA row tiles)
    constexpr int NCT = (CO/16 + 7) / 8;       // Co tiles per wave (8 waves)
    __shared__ float sW[MT * 144];             // 18 KB geometry weights (f32)
    __shared__ int   sN[MT * 9];
    __shared__ alignas(16) __bf16 sF[MT * 9 * 32];  // 18 KB relu'd neighbor features (bf16)
    __shared__ alignas(16) __bf16 sA[MT * 32];      // 2 KB A tiles

    const int tid  = threadIdx.x;
    const int wave = tid >> 5;
    const int lane = tid & 31;
    const int m    = lane & 15;                // A row / B,C,D column within tile
    const int half = lane >> 4;
    const int row0 = blockIdx.x * MT;

    for (int i = tid; i < MT * 144; i += 256) sW[i] = wfull[(size_t)row0 * 144 + i];
    for (int i = tid; i < MT * 9;   i += 256) sN[i] = nbr[(size_t)row0 * 9 + i];

    v8f zero = {0.f,0.f,0.f,0.f,0.f,0.f,0.f,0.f};
    v8f acc[2][NCT];
    #pragma unroll
    for (int rt = 0; rt < 2; ++rt)
        #pragma unroll
        for (int ct = 0; ct < NCT; ++ct) acc[rt][ct] = zero;

    for (int cc = 0; cc < CIN / 32; ++cc) {
        __syncthreads();
        // stage relu(features)->bf16 for 32 particles x 9 neighbors x 32-channel chunk,
        // prefetch the next chunk's rows while we are at it
        for (int i = tid; i < MT * 9 * 32; i += 256) {
            int r = i / 288; int rem = i - r * 288; int k = rem >> 5; int c = rem & 31;
            const float* src = &fin[(size_t)sN[r*9 + k] * CIN + cc*32 + c];
            if ((cc + 1 < CIN / 32) && (c == 0))
                __builtin_prefetch(src + 32, 0, 0);
            sF[i] = (__bf16)fmaxf(*src, 0.f);
        }
        __syncthreads();
        for (int g = 0; g < 16; ++g) {
            // gather: f32 accumulate over 9 neighbors (bf16 sources) -> bf16 A tiles
            for (int i = tid; i < MT * 32; i += 256) {
                int r = i >> 5; int c = i & 31;
                float t = 0.f;
                #pragma unroll
                for (int k = 0; k < 9; ++k)
                    t += sW[r*144 + k*16 + g] * (float)sF[(r*9 + k)*32 + c];
                sA[i] = (__bf16)t;
            }
            __syncthreads();
            // A fragments: lane m rows; K runs [16h..16h+15] (contiguous)
            union AU { struct { v8bf lo, hi; } p; v16bf v; } au;
            v16bf afrag[2];
            #pragma unroll
            for (int rt = 0; rt < 2; ++rt) {
                const __bf16* bp = &sA[(rt*16 + m) * 32 + 16 * half];
                au.p.lo = *(const v8bf*)bp;
                au.p.hi = *(const v8bf*)(bp + 8);
                afrag[rt] = au.v;
            }
            #pragma unroll
            for (int ct = 0; ct < NCT; ++ct) {
                int co = (wave + ct * 8) * 16;
                if (co < CO) {
                    // B fragment: lane holds column co+m, K = 16*half + j (contiguous bf16)
                    v16bf bfrag = *(const v16bf*)&wB[((size_t)g * CO + co + m) * CIN + cc*32 + 16*half];
                    #pragma unroll
                    for (int rt = 0; rt < 2; ++rt)
                        acc[rt][ct] = __builtin_amdgcn_wmma_f32_16x16x32_bf16(
                            false, afrag[rt], false, bfrag, (short)0, acc[rt][ct], false, false);
                }
            }
            __syncthreads();
        }
        // dense branch: A = relu(self feats) = staged neighbor k=4 (center == self), already bf16
        {
            union AU { struct { v8bf lo, hi; } p; v16bf v; } au;
            v16bf afrag[2];
            #pragma unroll
            for (int rt = 0; rt < 2; ++rt) {
                const __bf16* bp = &sF[((rt*16 + m)*9 + 4) * 32 + 16 * half];
                au.p.lo = *(const v8bf*)bp;
                au.p.hi = *(const v8bf*)(bp + 8);
                afrag[rt] = au.v;
            }
            #pragma unroll
            for (int ct = 0; ct < NCT; ++ct) {
                int co = (wave + ct * 8) * 16;
                if (co < CO) {
                    v16bf bfrag = *(const v16bf*)&wD[(size_t)(co + m) * CIN + cc*32 + 16*half];
                    #pragma unroll
                    for (int rt = 0; rt < 2; ++rt)
                        acc[rt][ct] = __builtin_amdgcn_wmma_f32_16x16x32_bf16(
                            false, afrag[rt], false, bfrag, (short)0, acc[rt][ct], false, false);
                }
            }
        }
    }
    // epilogue: D layout lane=n, m = vgpr + 8*half; add biases (+ residual)
    #pragma unroll
    for (int ct = 0; ct < NCT; ++ct) {
        int co = (wave + ct * 8) * 16;
        if (co < CO) {
            float bias = cb[co + m] + db[co + m];
            #pragma unroll
            for (int rt = 0; rt < 2; ++rt)
                #pragma unroll
                for (int r = 0; r < 8; ++r) {
                    size_t row = (size_t)(row0 + rt*16 + r + 8*half);
                    float v = acc[rt][ct][r] + bias;
                    if (RES) v += fin[row * CIN + co + m];
                    fout[row * CO + co + m] = v;
                }
        }
    }
}

// ---------------- layer 4 (Co=1): project-then-gather ----------------
__global__ void proj4_kernel(const float* __restrict__ f3, const float* __restrict__ w4,
                             const float* __restrict__ dw4, const float* __restrict__ db4,
                             float* __restrict__ Y, float* __restrict__ dsum, int N) {
    int n = blockIdx.x * blockDim.x + threadIdx.x;
    if (n >= N) return;
    float yg[16];
    #pragma unroll
    for (int g = 0; g < 16; ++g) yg[g] = 0.f;
    float ds = 0.f;
    for (int c = 0; c < 256; ++c) {
        float v = fmaxf(f3[(size_t)n*256 + c], 0.f);
        ds += v * dw4[c];
        #pragma unroll
        for (int g = 0; g < 16; ++g) yg[g] += v * w4[g*256 + c];
    }
    #pragma unroll
    for (int g = 0; g < 16; ++g) Y[(size_t)n*16 + g] = yg[g];
    dsum[n] = ds + db4[0];
}

__global__ void final4_kernel(const float* __restrict__ Y, const float* __restrict__ dsum,
                              const float* __restrict__ wfull, const int* __restrict__ nbr,
                              const float* __restrict__ cb4, float* __restrict__ out, int N) {
    int n = blockIdx.x * blockDim.x + threadIdx.x;
    if (n >= N) return;
    float s = cb4[0] + dsum[n];
    #pragma unroll
    for (int k = 0; k < 9; ++k) {
        int id = nbr[n*9 + k];
        #pragma unroll
        for (int g = 0; g < 16; ++g)
            s += wfull[(size_t)n*144 + k*16 + g] * Y[(size_t)id*16 + g];
    }
    out[n] = tanhf(s) * 0.8f + 1.f;
}

// ---------------- launcher ----------------
extern "C" void kernel_launch(void* const* d_in, const int* in_sizes, int n_in,
                              void* d_out, int out_size, void* d_ws, size_t ws_size,
                              hipStream_t stream) {
    const float* pos   = (const float*)d_in[0];
    const float* vel   = (const float*)d_in[1];
    const int*   nbr   = (const int*)d_in[2];
    const unsigned char* mask = (const unsigned char*)d_in[3];
    const float* gamma = (const float*)d_in[4];
    const float* beta  = (const float*)d_in[5];
    const float* cw0 = (const float*)d_in[6];  const float* cb0 = (const float*)d_in[7];
    const float* dw0 = (const float*)d_in[8];  const float* db0 = (const float*)d_in[9];
    const float* cw1 = (const float*)d_in[10]; const float* cb1 = (const float*)d_in[11];
    const float* dw1 = (const float*)d_in[12]; const float* db1 = (const float*)d_in[13];
    const float* cw2 = (const float*)d_in[14]; const float* cb2 = (const float*)d_in[15];
    const float* dw2 = (const float*)d_in[16]; const float* db2 = (const float*)d_in[17];
    const float* cw3 = (const float*)d_in[18]; const float* cb3 = (const float*)d_in[19];
    const float* dw3 = (const float*)d_in[20]; const float* db3 = (const float*)d_in[21];
    const float* cw4 = (const float*)d_in[22]; const float* cb4 = (const float*)d_in[23];
    const float* dw4 = (const float*)d_in[24]; const float* db4 = (const float*)d_in[25];
    float* out = (float*)d_out;
    const int N = in_sizes[0] / 2;
    (void)n_in; (void)out_size; (void)ws_size;

    char* ws = (char*)d_ws;
    size_t off = 0;
    auto alloc = [&](size_t bytes) -> char* {
        char* p = ws + off; off += (bytes + 255) & ~(size_t)255; return p;
    };
    float* stats   = (float*)alloc(4 * sizeof(float));
    float* featsin = (float*)alloc((size_t)N * 3 * 4);
    float* wfull   = (float*)alloc((size_t)N * 144 * 4);
    float* f0      = (float*)alloc((size_t)N * 64 * 4);
    float* f1      = (float*)alloc((size_t)N * 64 * 4);
    float* f2      = (float*)alloc((size_t)N * 128 * 4);
    float* f3      = (float*)alloc((size_t)N * 256 * 4);
    float* Y       = (float*)alloc((size_t)N * 16 * 4);
    float* dsum    = (float*)alloc((size_t)N * 4);
    __bf16* wB1 = (__bf16*)alloc((size_t)16*64*64*2);
    __bf16* wD1 = (__bf16*)alloc((size_t)64*64*2);
    __bf16* wB2 = (__bf16*)alloc((size_t)16*64*128*2);
    __bf16* wD2 = (__bf16*)alloc((size_t)64*128*2);
    __bf16* wB3 = (__bf16*)alloc((size_t)16*128*256*2);
    __bf16* wD3 = (__bf16*)alloc((size_t)128*256*2);

    stats_zero<<<1, 32, 0, stream>>>(stats);
    stats_acc<<<128, 256, 0, stream>>>(vel, stats, N);
    bn_feats<<<(N + 255)/256, 256, 0, stream>>>(vel, stats, gamma, beta, featsin, N);
    geom_kernel<<<(N*9 + 255)/256, 256, 0, stream>>>(pos, nbr, mask, wfull, N);

    w_to_bf16T<<<(16*64*64  + 255)/256, 256, 0, stream>>>(cw1, wB1, 16, 64, 64);
    w_to_bf16T<<<(64*64     + 255)/256, 256, 0, stream>>>(dw1, wD1, 1, 64, 64);
    w_to_bf16T<<<(16*64*128 + 255)/256, 256, 0, stream>>>(cw2, wB2, 16, 64, 128);
    w_to_bf16T<<<(64*128    + 255)/256, 256, 0, stream>>>(dw2, wD2, 1, 64, 128);
    w_to_bf16T<<<(16*128*256+ 255)/256, 256, 0, stream>>>(cw3, wB3, 16, 128, 256);
    w_to_bf16T<<<(128*256   + 255)/256, 256, 0, stream>>>(dw3, wD3, 1, 128, 256);

    layer0_kernel<<<(N + 255)/256, 256, 0, stream>>>(featsin, wfull, nbr, cw0, cb0, dw0, db0, f0, N);
    conv_layer_kernel<64, 64, true ><<<N/32, 256, 0, stream>>>(f0, wB1, cb1, wD1, db1, wfull, nbr, f1);
    conv_layer_kernel<64, 128,false><<<N/32, 256, 0, stream>>>(f1, wB2, cb2, wD2, db2, wfull, nbr, f2);
    conv_layer_kernel<128,256,false><<<N/32, 256, 0, stream>>>(f2, wB3, cb3, wD3, db3, wfull, nbr, f3);
    proj4_kernel<<<(N + 255)/256, 256, 0, stream>>>(f3, cw4, dw4, db4, Y, dsum, N);
    final4_kernel<<<(N + 255)/256, 256, 0, stream>>>(Y, dsum, wfull, nbr, cb4, out, N);
}
